// RNNSeq2Seq_60868276519614
// MI455X (gfx1250) — matrix operand order repaired and checked by reading, not compiled, not run on hardware
//
#include <hip/hip_runtime.h>
#include <hip/hip_bf16.h>

// ---------------- problem constants ----------------
#define Hh     1024
#define FOURH  4096
#define BATCH  32
#define TSEQ   128
#define VOCAB  32000

typedef __bf16 bf16_t;
typedef __attribute__((ext_vector_type(16))) __bf16 v16bf;
typedef __attribute__((ext_vector_type(8)))  __bf16 v8bf;
typedef __attribute__((ext_vector_type(8)))  float  v8f;

// ---------------- WMMA helpers (gfx1250, wave32) ----------------
// A fragment: 16x32 bf16 tile, row-major (row stride ld).
// Layout: lanes 0-15 -> M=lane, K 0..15; lanes 16-31 -> K 16..31 (matches B fragment).
__device__ __forceinline__ v16bf load_frag_a(const bf16_t* __restrict__ base, int ld,
                                             int row0, int k0, int lane) {
  int r  = row0 + (lane & 15);
  int ko = (lane & 16) ? 16 : 0;
  const bf16_t* p = base + (size_t)r * ld + k0 + ko;
  v8bf lo = *(const v8bf*)(p);
  v8bf hi = *(const v8bf*)(p + 8);
  v16bf a;
#pragma unroll
  for (int i = 0; i < 8; ++i) { a[i] = lo[i]; a[i + 8] = hi[i]; }
  return a;
}

// B fragment: 32x16 bf16 tile; weights stored TRANSPOSED as [N][K], so each
// lane's 16 K-values are contiguous: lanes 0-15 -> N=lane, K 0..15; lanes 16-31 -> K 16..31.
__device__ __forceinline__ v16bf load_frag_b(const bf16_t* __restrict__ baseT, int ld,
                                             int n, int k0, int lane) {
  const bf16_t* p = baseT + (size_t)n * ld + k0 + ((lane & 16) ? 16 : 0);
  return *(const v16bf*)(p);
}

__device__ __forceinline__ v8f wmma_bf16(v16bf a, v16bf b, v8f c) {
  return __builtin_amdgcn_wmma_f32_16x16x32_bf16(
      /*neg_a=*/false, a, /*neg_b=*/false, b,
      /*c_mod=*/(short)0, c, /*reuse_a=*/false, /*reuse_b=*/false);
}

// gfx1250 hardware transcendental: v_tanh_f32
__device__ __forceinline__ float fast_tanh(float x) { return __builtin_amdgcn_tanhf(x); }
// sigmoid(x) = 0.5*tanh(x/2) + 0.5  -> single v_tanh_f32 + fma
__device__ __forceinline__ float sigf(float x) { return fmaf(0.5f, fast_tanh(0.5f * x), 0.5f); }

// ---------------- init: zero h/c state and barrier counters ----------------
__global__ void init_state_k(bf16_t* __restrict__ Hbuf, float* __restrict__ Cbuf,
                             unsigned* __restrict__ cnt) {
  int i = blockIdx.x * blockDim.x + threadIdx.x;
  if (i < 2 * BATCH * Hh) Hbuf[i] = (bf16_t)0.f;
  if (i < BATCH * Hh)     Cbuf[i] = 0.f;
  if (i < 8)              cnt[i]  = 0u;
}

// ---------------- embedding gather -> bf16, [T,B,H] layout ----------------
__global__ void embed_k(const int* __restrict__ tok, const float* __restrict__ emb,
                        bf16_t* __restrict__ X) {
  int r = blockIdx.x;                // r = t*B + b
  int t = r / BATCH, b = r - t * BATCH;
  int id = tok[b * TSEQ + t];
  const float* src = emb + (size_t)id * Hh;
  bf16_t* dst = X + (size_t)r * Hh;
  for (int k = threadIdx.x; k < Hh; k += blockDim.x) dst[k] = (bf16_t)src[k];
}

// ---------------- fp32 [K][N] -> bf16 transposed [N][K] ----------------
__global__ void convT_k(const float* __restrict__ Win, bf16_t* __restrict__ WT,
                        int K, int N) {
  size_t idx = (size_t)blockIdx.x * blockDim.x + threadIdx.x;
  size_t tot = (size_t)K * N;
  if (idx >= tot) return;
  size_t n = idx / K, k = idx - n * K;
  WT[idx] = (bf16_t)Win[k * (size_t)N + n];
}

// ---------------- generic bf16 WMMA GEMM: Z[M,N] = X[M,K] * WT[N,K]^T + bias ----
// Workgroup = 4 waves; wave covers 32(M) x 64(N); WG tile = 32 x 256.
__global__ __launch_bounds__(128)
void gemm_bf16_k(const bf16_t* __restrict__ X, const bf16_t* __restrict__ WT,
                 const float* __restrict__ bias, float* __restrict__ Z,
                 int N, int K) {
  const int lane = threadIdx.x & 31;
  const int wave = threadIdx.x >> 5;
  const int m0 = blockIdx.y * 32;
  const int n0 = blockIdx.x * 256 + wave * 64;
  v8f acc[2][4] = {};
  for (int k0 = 0; k0 < K; k0 += 32) {
    v16bf a0 = load_frag_a(X, K, m0,      k0, lane);
    v16bf a1 = load_frag_a(X, K, m0 + 16, k0, lane);
#pragma unroll
    for (int tt = 0; tt < 4; ++tt) {
      int n = n0 + tt * 16 + (lane & 15);
      // gfx1250 prefetch of next K-slab of this weight row (global_prefetch_b8)
      __builtin_prefetch((const void*)(WT + (size_t)n * K + k0 + 32), 0, 1);
      v16bf b = load_frag_b(WT, K, n, k0, lane);
      acc[0][tt] = wmma_bf16(a0, b, acc[0][tt]);
      acc[1][tt] = wmma_bf16(a1, b, acc[1][tt]);
    }
  }
#pragma unroll
  for (int mt = 0; mt < 2; ++mt)
#pragma unroll
    for (int tt = 0; tt < 4; ++tt) {
      int n = n0 + tt * 16 + (lane & 15);
      float bv = bias ? bias[n] : 0.f;
#pragma unroll
      for (int v = 0; v < 8; ++v) {
        int m = m0 + mt * 16 + ((lane >> 4) << 3) + v;
        Z[(size_t)m * N + n] = acc[mt][tt][v] + bv;
      }
    }
}

// ---------------- persistent LSTM scan over T steps ----------------
// 32 WGs; WG g owns hidden units [g*32, g*32+32). Per step the 32x1024 bf16 h
// matrix is staged global->LDS with gfx1250 async-to-LDS copies (ASYNCcnt),
// then A fragments come from LDS (ds_load) while U streams from L2.
__global__ __launch_bounds__(128)
void lstm_scan_k(const float* __restrict__ XZ,  // [T*B, 4H] fp32 (bias included)
                 const bf16_t* __restrict__ UT, // [4H][H] bf16 transposed
                 bf16_t* __restrict__ XN,       // [T*B, H] bf16: output sequence
                 bf16_t* __restrict__ Hbuf,     // [2][B][H] bf16 double buffer
                 float* __restrict__ Cbuf,      // [B][H] fp32 carry
                 unsigned* __restrict__ cnt,    // grid barrier counter (zeroed)
                 int nsteps) {
  const int tid = threadIdx.x, lane = tid & 31, wave = tid >> 5;
  const int u0 = blockIdx.x * 32;
  const unsigned nwg = gridDim.x;
  __shared__ bf16_t hsh[BATCH * Hh];   // 64 KB staged h
  __shared__ float  zld[BATCH * 128];  // 16 KB z slice

  const unsigned hsh_lds = (unsigned)(uintptr_t)(&hsh[0]);  // LDS byte offset (flat low 32 bits)

  float creg[8];
#pragma unroll
  for (int e = 0; e < 8; ++e) {
    int p = tid + e * 128; int b = p >> 5; int u = u0 + (p & 31);
    creg[e] = Cbuf[b * Hh + u];
  }

  for (int t = 0; t < nsteps; ++t) {
    const bf16_t* hc = Hbuf + (size_t)(t & 1) * BATCH * Hh;
    bf16_t*       hn = Hbuf + (size_t)((t + 1) & 1) * BATCH * Hh;
    const float* xzt = XZ + (size_t)t * BATCH * FOURH;

    // ---- async stage h (64 KB) global -> LDS: 32 rounds x 128 lanes x 16 B ----
    {
      const char* g = (const char*)hc;
#pragma unroll 4
      for (int it = 0; it < (BATCH * Hh * 2) / (128 * 16); ++it) {
        int off = (it * 128 + tid) * 16;
        unsigned ldsa = hsh_lds + (unsigned)off;
        const void* ga = (const void*)(g + off);
        asm volatile("global_load_async_to_lds_b128 %0, %1, off"
                     :: "v"(ldsa), "v"(ga) : "memory");
      }
      asm volatile("s_wait_asynccnt 0x0" ::: "memory");
      __syncthreads();
    }

    v8f acc[2][2] = {};
    for (int k0 = 0; k0 < Hh; k0 += 32) {
      v16bf a0 = load_frag_a(hsh, Hh, 0,  k0, lane);
      v16bf a1 = load_frag_a(hsh, Hh, 16, k0, lane);
#pragma unroll
      for (int tt = 0; tt < 2; ++tt) {
        int lcol = (2 * wave + tt) * 16 + (lane & 15);     // [w*32, w*32+32)
        int n = wave * Hh + u0 + (lcol & 31);              // row of UT
        __builtin_prefetch((const void*)(UT + (size_t)n * Hh + k0 + 32), 0, 1);
        v16bf b = load_frag_b(UT, Hh, n, k0, lane);
        acc[0][tt] = wmma_bf16(a0, b, acc[0][tt]);
        acc[1][tt] = wmma_bf16(a1, b, acc[1][tt]);
      }
    }

    // z slice (32 batch x 128 gate-cols) -> LDS, adding precomputed xz
#pragma unroll
    for (int mt = 0; mt < 2; ++mt)
#pragma unroll
      for (int tt = 0; tt < 2; ++tt) {
        int lcol = (2 * wave + tt) * 16 + (lane & 15);
        int ncol = wave * Hh + u0 + (lcol & 31);
#pragma unroll
        for (int v = 0; v < 8; ++v) {
          int b = mt * 16 + ((lane >> 4) << 3) + v;
          zld[b * 128 + lcol] = acc[mt][tt][v] + xzt[(size_t)b * FOURH + ncol];
        }
      }
    __syncthreads();

    // gates: thread owns 8 fixed (b,u) pairs -> c stays in registers all T steps
#pragma unroll
    for (int e = 0; e < 8; ++e) {
      int p = tid + e * 128; int b = p >> 5; int j = p & 31; int u = u0 + j;
      float zi = zld[b * 128 +  0 + j];
      float zf = zld[b * 128 + 32 + j];
      float zg = zld[b * 128 + 64 + j];
      float zo = zld[b * 128 + 96 + j];
      float cn = sigf(zf) * creg[e] + sigf(zi) * fast_tanh(zg);
      creg[e] = cn;
      float hv = sigf(zo) * fast_tanh(cn);
      bf16_t hb = (bf16_t)hv;
      hn[b * Hh + u] = hb;
      XN[((size_t)t * BATCH + b) * Hh + u] = hb;
    }
    __syncthreads();

    // device-scope grid barrier (monotonic counter; zeroed by init each call)
    if (tid == 0) {
      __threadfence();
      __hip_atomic_fetch_add(cnt, 1u, __ATOMIC_RELEASE, __HIP_MEMORY_SCOPE_AGENT);
    }
    unsigned target = (unsigned)(t + 1) * nwg;
    while (__hip_atomic_load(cnt, __ATOMIC_ACQUIRE, __HIP_MEMORY_SCOPE_AGENT) < target)
      __builtin_amdgcn_s_sleep(2);
    __syncthreads();
  }

#pragma unroll
  for (int e = 0; e < 8; ++e) {
    int p = tid + e * 128; int b = p >> 5; int u = u0 + (p & 31);
    Cbuf[b * Hh + u] = creg[e];
  }
}

// ---------------- host-side orchestration ----------------
extern "C" void kernel_launch(void* const* d_in, const int* in_sizes, int n_in,
                              void* d_out, int out_size, void* d_ws, size_t ws_size,
                              hipStream_t stream) {
  const int*   enc_tok = (const int*)d_in[0];
  const int*   dec_tok = (const int*)d_in[1];
  const float* emb     = (const float*)d_in[2];
  const float* enc_W   = (const float*)d_in[3];
  const float* enc_U   = (const float*)d_in[4];
  const float* enc_b   = (const float*)d_in[5];
  const float* dec_W   = (const float*)d_in[6];
  const float* dec_U   = (const float*)d_in[7];
  const float* dec_b   = (const float*)d_in[8];
  const float* dense_W = (const float*)d_in[9];
  const float* dense_b = (const float*)d_in[10];
  float* out = (float*)d_out;

  // workspace layout (bytes)
  char* ws = (char*)d_ws;
  bf16_t*   X   = (bf16_t*)(ws);                            // [4096,1024] bf16  (8 MiB)
  float*    XZ  = (float*) (ws + (8ull  << 20));            // [4096,4096] f32   (64 MiB)
  bf16_t*   WT  = (bf16_t*)(ws + (72ull << 20));            // [4096,1024] bf16  (8 MiB)
  bf16_t*   UT  = (bf16_t*)(ws + (80ull << 20));            // [4096,1024] bf16  (8 MiB)
  bf16_t*   Hb  = (bf16_t*)(ws + (88ull << 20));            // [2,32,1024] bf16
  float*    Cb  = (float*) (ws + (88ull << 20) + 131072);   // [32,1024] f32
  unsigned* Cnt = (unsigned*)(ws + (88ull << 20) + 262144); // 8 barrier counters
  bf16_t*   DWT = (bf16_t*)(ws + (8ull  << 20));            // reuse XZ region: [32000,1024] bf16

  const int TB = TSEQ * BATCH;                 // 4096 sequence rows

  init_state_k<<<(2 * BATCH * Hh + 255) / 256, 256, 0, stream>>>(Hb, Cb, Cnt);

  // ---- encoder ----
  embed_k<<<TB, 256, 0, stream>>>(enc_tok, emb, X);
  for (int l = 0; l < 2; ++l) {
    int wgs = (Hh * FOURH + 255) / 256;
    convT_k<<<wgs, 256, 0, stream>>>(enc_W + (size_t)l * Hh * FOURH, WT, Hh, FOURH);
    convT_k<<<wgs, 256, 0, stream>>>(enc_U + (size_t)l * Hh * FOURH, UT, Hh, FOURH);
    dim3 g(FOURH / 256, TB / 32);
    gemm_bf16_k<<<g, 128, 0, stream>>>(X, WT, enc_b + (size_t)l * FOURH, XZ, FOURH, Hh);
    lstm_scan_k<<<Hh / 32, 128, 0, stream>>>(XZ, UT, X, Hb, Cb, Cnt + l, TSEQ);
  }

  // ---- decoder ----
  embed_k<<<TB, 256, 0, stream>>>(dec_tok, emb, X);
  for (int l = 0; l < 2; ++l) {
    int wgs = (Hh * FOURH + 255) / 256;
    convT_k<<<wgs, 256, 0, stream>>>(dec_W + (size_t)l * Hh * FOURH, WT, Hh, FOURH);
    convT_k<<<wgs, 256, 0, stream>>>(dec_U + (size_t)l * Hh * FOURH, UT, Hh, FOURH);
    dim3 g(FOURH / 256, TB / 32);
    gemm_bf16_k<<<g, 128, 0, stream>>>(X, WT, dec_b + (size_t)l * FOURH, XZ, FOURH, Hh);
    lstm_scan_k<<<Hh / 32, 128, 0, stream>>>(XZ, UT, X, Hb, Cb, Cnt + 2 + l, TSEQ);
  }

  // ---- dense head: out[32,32000] = h_final @ dense_W + dense_b ----
  // final h lives in Hb buffer 0 (T=128 even)
  {
    int wgs = (Hh * VOCAB + 255) / 256;
    convT_k<<<wgs, 256, 0, stream>>>(dense_W, DWT, Hh, VOCAB);
    dim3 g(VOCAB / 256, 1);
    gemm_bf16_k<<<g, 128, 0, stream>>>(Hb, DWT, dense_b, out, VOCAB, Hh);
  }
}
